// SiglipAttention_24790551233355
// MI455X (gfx1250) — matrix operand order repaired and checked by reading, not compiled, not run on hardware
//
#include <hip/hip_runtime.h>

// ---------- types ----------
typedef __attribute__((ext_vector_type(16))) __bf16          v16bf;
typedef __attribute__((ext_vector_type(8)))  float           v8f;
typedef __attribute__((ext_vector_type(4)))  unsigned int    u32x4;
typedef __attribute__((ext_vector_type(4)))  float           f32x4;
typedef __attribute__((ext_vector_type(4)))  unsigned short  u16x4;

#define DEV __device__ __forceinline__

// ---------- problem constants ----------
#define BB    32
#define SS    729
#define HH    1152
#define NH    16
#define HD    72
#define N3    3456
#define MTOT  23328      // BB*SS
#define BHT   512        // BB*NH
#define SPAD  736        // S padded to 16
#define DQK   96         // head_dim padded to 3*32 for QK^T K-steps
#define DV    80         // head_dim padded to 5*16 for PV N-tiles
#define LOG2E 1.44269504088896340736f
#define QSCALE 0.11785113019775793f   // 72^-0.5

// ---------- fragment union (A/B operands: 16 bf16 = 32 bytes) ----------
union FragAB { v16bf v; u32x4 q[2]; unsigned short h[16]; };

DEV unsigned short f2bf(float f) {
  union { float f; unsigned u; } x; x.f = f;
  unsigned u = x.u;
  u += 0x7fffu + ((u >> 16) & 1u);           // round-to-nearest-even
  return (unsigned short)(u >> 16);
}

// ---------- fp32 -> bf16 convert (vectorized x4) ----------
__global__ __launch_bounds__(256)
void k_f32_to_bf16(const float* __restrict__ src, unsigned short* __restrict__ dst, int n4) {
  int i = blockIdx.x * blockDim.x + threadIdx.x;
  if (i >= n4) return;
  f32x4 f = ((const f32x4*)src)[i];
  u16x4 o;
  o[0] = f2bf(f[0]); o[1] = f2bf(f[1]); o[2] = f2bf(f[2]); o[3] = f2bf(f[3]);
  ((u16x4*)dst)[i] = o;
}

// ---------- QKV GEMM: (MTOT x HH) @ (N3 x HH)^T, scatter into Q/K/V^T ----------
// block: 256 threads = 8 waves arranged 4(M) x 2(N); wave tile 64x64 (4x4 WMMA tiles)
// => 16 WMMAs per 16 b128 loads per k-step (1:1 issue ratio)
__global__ __launch_bounds__(256)
void k_gemm_qkv(const unsigned short* __restrict__ Xb,
                const unsigned short* __restrict__ Wb,
                const float* __restrict__ bias,
                unsigned short* __restrict__ Qb,   // (BHT, SPAD, DQK)
                unsigned short* __restrict__ Kb,   // (BHT, SPAD, DQK)
                unsigned short* __restrict__ Vt)   // (BHT, DV, SPAD)
{
  const int lane = threadIdx.x & 31;
  const int wid  = threadIdx.x >> 5;
  const int hs = lane >> 4, l15 = lane & 15;
  const int waveM = wid >> 1, waveN = wid & 1;
  const int mBlk = blockIdx.y * 256 + waveM * 64;
  const int nBlk = blockIdx.x * 128 + waveN * 64;

  const unsigned short* aptr[4];
#pragma unroll
  for (int im = 0; im < 4; ++im) {
    int m = mBlk + im * 16 + l15;
    if (m >= MTOT) m = MTOT - 1;               // clamp edge loads
    aptr[im] = Xb + (size_t)m * HH;
  }
  const unsigned short* bptr[4];
#pragma unroll
  for (int in = 0; in < 4; ++in)
    bptr[in] = Wb + (size_t)(nBlk + in * 16 + l15) * HH;

  v8f acc[4][4];
#pragma unroll
  for (int im = 0; im < 4; ++im)
#pragma unroll
    for (int in = 0; in < 4; ++in)
#pragma unroll
      for (int v = 0; v < 8; ++v) acc[im][in][v] = 0.0f;

  for (int k0 = 0; k0 < HH; k0 += 32) {
    FragAB a[4], b[4];
#pragma unroll
    for (int im = 0; im < 4; ++im) {
      const unsigned short* p = aptr[im] + k0 + 8 * hs;   // A: e0-7 -> K=8*hs.., e8-15 -> K=16+8*hs..
      a[im].q[0] = *(const u32x4*)(p);
      a[im].q[1] = *(const u32x4*)(p + 16);
    }
#pragma unroll
    for (int in = 0; in < 4; ++in) {
      const unsigned short* p = bptr[in] + k0 + 8 * hs;   // B: e0-7 -> K=8*hs.., e8-15 -> K=16+8*hs..
      b[in].q[0] = *(const u32x4*)(p);
      b[in].q[1] = *(const u32x4*)(p + 16);
    }
#pragma unroll
    for (int im = 0; im < 4; ++im)
#pragma unroll
      for (int in = 0; in < 4; ++in)
        acc[im][in] = __builtin_amdgcn_wmma_f32_16x16x32_bf16(
            false, a[im].v, false, b[in].v, (short)0, acc[im][in], false, false);
  }

  // epilogue: bias, fold softmax scale into Q, scatter to attention layouts
#pragma unroll
  for (int in = 0; in < 4; ++in) {
    int n = nBlk + in * 16 + l15;
    float bs = bias[n];
    int t = n / HH;
    int r = n - t * HH;
    int h = r / HD;
    int d = r - h * HD;
#pragma unroll
    for (int im = 0; im < 4; ++im) {
#pragma unroll
      for (int v = 0; v < 8; ++v) {
        int m = mBlk + im * 16 + v + 8 * hs;   // D layout: row = v + 8*(lane>=16)
        if (m >= MTOT) continue;
        float val = acc[im][in][v] + bs;
        int bI = m / SS;
        int s  = m - bI * SS;
        int bh = bI * NH + h;
        if (t == 0) {
          Qb[((size_t)bh * SPAD + s) * DQK + d] = f2bf(val * QSCALE);
        } else if (t == 1) {
          Kb[((size_t)bh * SPAD + s) * DQK + d] = f2bf(val);
        } else {
          Vt[((size_t)bh * DV + d) * SPAD + s] = f2bf(val);
        }
      }
    }
  }
}

// ---------- Flash attention: 1 wave = 16 query rows, keys in chunks of 32 ----------
__global__ __launch_bounds__(128)
void k_flash(const unsigned short* __restrict__ Qb,
             const unsigned short* __restrict__ Kb,
             const unsigned short* __restrict__ Vt,
             unsigned short* __restrict__ Ctx)   // (MTOT, HH) bf16
{
  __shared__ unsigned short lds[4 * 16 * 32];    // 1KB P-transpose tile per wave
  const int lane = threadIdx.x & 31;
  const int wid  = threadIdx.x >> 5;
  const int hs = lane >> 4, l15 = lane & 15;
  const int bh = blockIdx.x;
  const int qbase = blockIdx.y * 64 + wid * 16;
  unsigned short* myLds = lds + wid * 512;

  // Q fragments: 3 K-chunks of 32 over padded head dim 96 (scale pre-folded)
  FragAB fq[3];
  {
    int qr = qbase + l15; if (qr > SS - 1) qr = SS - 1;
    const unsigned short* qp = Qb + ((size_t)bh * SPAD + qr) * DQK;
#pragma unroll
    for (int c = 0; c < 3; ++c) {
      const unsigned short* p = qp + c * 32 + 8 * hs;
      fq[c].q[0] = *(const u32x4*)(p);
      fq[c].q[1] = *(const u32x4*)(p + 16);
    }
  }

  float mrow[8], lrow[8];
  v8f o[5];
#pragma unroll
  for (int v = 0; v < 8; ++v) { mrow[v] = -3.0e38f; lrow[v] = 0.0f; }
#pragma unroll
  for (int t = 0; t < 5; ++t)
#pragma unroll
    for (int v = 0; v < 8; ++v) o[t][v] = 0.0f;

  for (int kk = 0; kk < SPAD; kk += 32) {
    // ---- scores: two 16-key tiles, 3 WMMAs each over head dim ----
    v8f sc[2];
#pragma unroll
    for (int j = 0; j < 2; ++j) {
#pragma unroll
      for (int v = 0; v < 8; ++v) sc[j][v] = 0.0f;
      int kr = kk + j * 16 + l15;
      const unsigned short* kp = Kb + ((size_t)bh * SPAD + kr) * DQK;
#pragma unroll
      for (int c = 0; c < 3; ++c) {
        FragAB fb;
        const unsigned short* p = kp + c * 32 + 8 * hs;
        fb.q[0] = *(const u32x4*)(p);
        fb.q[1] = *(const u32x4*)(p + 16);
        sc[j] = __builtin_amdgcn_wmma_f32_16x16x32_bf16(
            false, fq[c].v, false, fb.v, (short)0, sc[j], false, false);
      }
      if (kr >= SS) {                            // mask padded keys
#pragma unroll
        for (int v = 0; v < 8; ++v) sc[j][v] = -1.0e30f;
      }
    }

    // ---- online softmax: row reductions across 16-lane groups ----
    float mnew[8], alpha[8];
#pragma unroll
    for (int v = 0; v < 8; ++v) {
      float mx = fmaxf(sc[0][v], sc[1][v]);
#pragma unroll
      for (int off = 1; off < 16; off <<= 1)
        mx = fmaxf(mx, __shfl_xor(mx, off, 32));
      mnew[v] = fmaxf(mrow[v], mx);
      alpha[v] = __builtin_exp2f((mrow[v] - mnew[v]) * LOG2E);
      mrow[v] = mnew[v];
    }
#pragma unroll
    for (int v = 0; v < 8; ++v) {
      float p0 = __builtin_exp2f((sc[0][v] - mnew[v]) * LOG2E);
      float p1 = __builtin_exp2f((sc[1][v] - mnew[v]) * LOG2E);
      float rs = p0 + p1;
#pragma unroll
      for (int off = 1; off < 16; off <<= 1)
        rs += __shfl_xor(rs, off, 32);
      lrow[v] = lrow[v] * alpha[v] + rs;
      int row = v + 8 * hs;                      // D-layout -> row-major 16x32 tile in LDS
      myLds[row * 32 + l15]      = f2bf(p0);
      myLds[row * 32 + 16 + l15] = f2bf(p1);
    }
#pragma unroll
    for (int t = 0; t < 5; ++t)
#pragma unroll
      for (int v = 0; v < 8; ++v) o[t][v] *= alpha[v];

    // ---- P as A-fragment from LDS (transpose via LDS roundtrip) ----
    FragAB fp;
    {
      const unsigned short* p = myLds + l15 * 32 + 8 * hs;
      fp.q[0] = *(const u32x4*)(p);
      fp.q[1] = *(const u32x4*)(p + 16);
    }
    // ---- O += P @ V : V^T gives contiguous B-fragment loads ----
#pragma unroll
    for (int t = 0; t < 5; ++t) {
      FragAB fv;
      const unsigned short* p = Vt + ((size_t)bh * DV + t * 16 + l15) * SPAD + kk + 8 * hs;
      fv.q[0] = *(const u32x4*)(p);
      fv.q[1] = *(const u32x4*)(p + 16);
      o[t] = __builtin_amdgcn_wmma_f32_16x16x32_bf16(
          false, fp.v, false, fv.v, (short)0, o[t], false, false);
    }
  }

  // ---- normalize and scatter context back to (B,S,H) bf16 ----
  float inv[8];
#pragma unroll
  for (int v = 0; v < 8; ++v) inv[v] = 1.0f / lrow[v];
  int bI = bh >> 4, h = bh & 15;
#pragma unroll
  for (int t = 0; t < 5; ++t) {
    int d = t * 16 + l15;
    if (d >= HD) continue;
#pragma unroll
    for (int v = 0; v < 8; ++v) {
      int s = qbase + v + 8 * hs;
      if (s >= SS) continue;
      Ctx[((size_t)(bI * SS + s)) * HH + h * HD + d] = f2bf(o[t][v] * inv[v]);
    }
  }
}

// ---------- Proj GEMM: (MTOT x HH) @ (HH x HH)^T + bias -> fp32 out ----------
__global__ __launch_bounds__(256)
void k_gemm_proj(const unsigned short* __restrict__ Ab,
                 const unsigned short* __restrict__ Wb,
                 const float* __restrict__ bias,
                 float* __restrict__ out)
{
  const int lane = threadIdx.x & 31;
  const int wid  = threadIdx.x >> 5;
  const int hs = lane >> 4, l15 = lane & 15;
  const int waveM = wid >> 1, waveN = wid & 1;
  const int mBlk = blockIdx.y * 256 + waveM * 64;
  const int nBlk = blockIdx.x * 128 + waveN * 64;

  const unsigned short* aptr[4];
#pragma unroll
  for (int im = 0; im < 4; ++im) {
    int m = mBlk + im * 16 + l15;
    if (m >= MTOT) m = MTOT - 1;
    aptr[im] = Ab + (size_t)m * HH;
  }
  const unsigned short* bptr[4];
#pragma unroll
  for (int in = 0; in < 4; ++in)
    bptr[in] = Wb + (size_t)(nBlk + in * 16 + l15) * HH;

  v8f acc[4][4];
#pragma unroll
  for (int im = 0; im < 4; ++im)
#pragma unroll
    for (int in = 0; in < 4; ++in)
#pragma unroll
      for (int v = 0; v < 8; ++v) acc[im][in][v] = 0.0f;

  for (int k0 = 0; k0 < HH; k0 += 32) {
    FragAB a[4], b[4];
#pragma unroll
    for (int im = 0; im < 4; ++im) {
      const unsigned short* p = aptr[im] + k0 + 8 * hs;
      a[im].q[0] = *(const u32x4*)(p);
      a[im].q[1] = *(const u32x4*)(p + 16);
    }
#pragma unroll
    for (int in = 0; in < 4; ++in) {
      const unsigned short* p = bptr[in] + k0 + 8 * hs;
      b[in].q[0] = *(const u32x4*)(p);
      b[in].q[1] = *(const u32x4*)(p + 16);
    }
#pragma unroll
    for (int im = 0; im < 4; ++im)
#pragma unroll
      for (int in = 0; in < 4; ++in)
        acc[im][in] = __builtin_amdgcn_wmma_f32_16x16x32_bf16(
            false, a[im].v, false, b[in].v, (short)0, acc[im][in], false, false);
  }

#pragma unroll
  for (int in = 0; in < 4; ++in) {
    int n = nBlk + in * 16 + l15;
    float bs = bias[n];
#pragma unroll
    for (int im = 0; im < 4; ++im)
#pragma unroll
      for (int v = 0; v < 8; ++v) {
        int m = mBlk + im * 16 + v + 8 * hs;
        if (m >= MTOT) continue;
        out[(size_t)m * HH + n] = acc[im][in][v] + bs;
      }
  }
}

// ---------- workspace layout (bytes) ----------
#define OFF_WQKV  0u
#define OFF_WPROJ 7962624u            // 3456*1152*2
#define OFF_XB    10616832u           // + 1152*1152*2
#define OFF_Q     64364544u           // + 23328*1152*2
#define OFF_K     136716288u          // + 512*736*96*2
#define OFF_VT    209068032u          // + 512*736*96*2
#define OFF_CTX   269361152u          // + 512*80*736*2

extern "C" void kernel_launch(void* const* d_in, const int* in_sizes, int n_in,
                              void* d_out, int out_size, void* d_ws, size_t ws_size,
                              hipStream_t stream) {
  (void)in_sizes; (void)n_in; (void)out_size; (void)ws_size;
  const float* X  = (const float*)d_in[0];
  const float* Wq = (const float*)d_in[1];
  const float* bq = (const float*)d_in[2];
  const float* Wp = (const float*)d_in[3];
  const float* bp = (const float*)d_in[4];

  char* ws = (char*)d_ws;
  unsigned short* wqkv  = (unsigned short*)(ws + OFF_WQKV);
  unsigned short* wproj = (unsigned short*)(ws + OFF_WPROJ);
  unsigned short* xb    = (unsigned short*)(ws + OFF_XB);
  unsigned short* qb    = (unsigned short*)(ws + OFF_Q);
  unsigned short* kb    = (unsigned short*)(ws + OFF_K);
  unsigned short* vt    = (unsigned short*)(ws + OFF_VT);
  unsigned short* ctx   = (unsigned short*)(ws + OFF_CTX);

  { int n4 = (N3 * HH) / 4;
    k_f32_to_bf16<<<(n4 + 255) / 256, 256, 0, stream>>>(Wq, wqkv, n4); }
  { int n4 = (HH * HH) / 4;
    k_f32_to_bf16<<<(n4 + 255) / 256, 256, 0, stream>>>(Wp, wproj, n4); }
  { int n4 = (MTOT * HH) / 4;
    k_f32_to_bf16<<<(n4 + 255) / 256, 256, 0, stream>>>(X, xb, n4); }

  // zero Q/K/V^T (covers head-dim and sequence padding regions)
  hipMemsetAsync(ws + OFF_Q, 0, (size_t)(OFF_CTX - OFF_Q), stream);

  k_gemm_qkv<<<dim3(N3 / 128, (MTOT + 255) / 256), 256, 0, stream>>>(xb, wqkv, bq, qb, kb, vt);
  k_flash  <<<dim3(BHT, (SS + 63) / 64), 128, 0, stream>>>(qb, kb, vt, ctx);
  k_gemm_proj<<<dim3(HH / 128, (MTOT + 255) / 256), 256, 0, stream>>>(ctx, wproj, bp, (float*)d_out);
}
